// FNO2D_14001593385570
// MI455X (gfx1250) — compile-verified, matrix operand order checked by reading, not run on hardware
//
#include <hip/hip_runtime.h>
#include <hip/hip_bf16.h>
#include <math.h>

// ---------------- constants -------------------------------------------------
#define BATCH   16
#define CHN     64
#define H0      256
#define W0      256
#define HP      264
#define WP      264
#define NP      (HP*WP)          // 69696
#define NMODE_Y 32               // ky 0..15 and 248..263
#define NMODE_X 16
#define NLAYER  4
#define TWO_PI  6.283185307179586f

typedef __attribute__((ext_vector_type(16))) __bf16 v16bf;
typedef __attribute__((ext_vector_type(8)))  float  v8f;

#if defined(__has_builtin)
#if __has_builtin(__builtin_amdgcn_tensor_load_to_lds)
#define HAVE_TDM 1
#endif
#endif

#ifdef HAVE_TDM
typedef __attribute__((ext_vector_type(4))) unsigned u32x4;
typedef __attribute__((ext_vector_type(8))) int      i32x8;
typedef __attribute__((ext_vector_type(4))) int      i32x4;
#endif

// ---------------- small helpers ---------------------------------------------
__device__ __forceinline__ float gelu_f(float x) {
    return 0.5f * x * (1.0f + erff(x * 0.70710678118654752f));
}

// async copy 16B global -> LDS (GLOBAL_LOAD_ASYNC_TO_LDS_B128, ASYNCcnt)
__device__ __forceinline__ void async_b128(unsigned lds_off, const void* gptr) {
    unsigned long long ga = (unsigned long long)(size_t)gptr;
    asm volatile("global_load_async_to_lds_b128 %0, %1, off"
                 :: "v"(lds_off), "v"(ga) : "memory");
}
__device__ __forceinline__ void wait_async0() {
    asm volatile("s_wait_asynccnt 0" ::: "memory");
}
__device__ __forceinline__ void wait_tensor0() {
#if defined(__has_builtin) && __has_builtin(__builtin_amdgcn_s_wait_tensorcnt)
    __builtin_amdgcn_s_wait_tensorcnt((short)0);
#else
    asm volatile("s_wait_tensorcnt 0" ::: "memory");
#endif
}

// ---- fragment-major ("swizzled") tile layout --------------------------------
// A 16-row x K tile is stored so that each (kstep, lane) fragment slice is one
// contiguous 32B block: FragU index = (ks*16 + m)*2 + half. A WMMA operand
// load is then two ds_load_b128, no per-element address math.
union FragU { unsigned u[8]; v16bf v; };
union B16x8 { uint4 u; __bf16 h[8]; };

__device__ __forceinline__ int swz_idx(int m, int k) {  // element index
    int ks  = k >> 5;
    int kk  = k & 31;
    int h   = (kk >> 3) & 1;
    int d   = ((kk >> 4) & 1) * 4 + ((kk >> 1) & 3);
    int pos = kk & 1;
    return (((ks * 16 + m) * 2 + h) * 8 + d) * 2 + pos;
}
__device__ __forceinline__ v16bf load_frag_swz(const __bf16* subtile, int ks, int lane) {
    const FragU* p = (const FragU*)subtile + ((ks * 16 + (lane & 15)) * 2 + (lane >> 4));
    return p->v;                          // 32B contiguous -> 2x ds_load_b128
}

// ---------------- lift + zero-pad (f32 in -> bf16 act), 8-wide ---------------
__global__ __launch_bounds__(256)
void k_lift(const float* __restrict__ x, const float* __restrict__ w,
            const float* __restrict__ bias, __bf16* __restrict__ h) {
    size_t idx = (size_t)blockIdx.x * 256 + threadIdx.x;
    size_t i8 = idx * 8;
    size_t t = i8;
    int px0 = (int)(t % NP); t /= NP;
    int c   = (int)(t % CHN);
    int b   = (int)(t / CHN);
    int y = px0 / WP, x0 = px0 - y * WP;       // 8-group never crosses a row
    B16x8 v;
    float wc = w[c], bc = bias[c];
    const float* xr = x + ((size_t)b * H0 + y) * W0;
    #pragma unroll
    for (int j = 0; j < 8; ++j) {
        int xx = x0 + j;
        float vv = (y < H0 && xx < W0) ? (xr[xx] * wc + bc) : 0.0f;
        v.h[j] = (__bf16)vv;
    }
    *(uint4*)(h + i8) = v.u;
}

// ---------------- forward partial DFT along W (16 kx modes) ------------------
__global__ __launch_bounds__(256)
void k_dftw(const __bf16* __restrict__ h, float* __restrict__ X1) {
    __shared__ __align__(16) float twr[16 * WP];
    __shared__ __align__(16) float twi[16 * WP];
    __shared__ __align__(16) __bf16 rowbuf[8][WP];
    int t = threadIdx.x;
    for (int i = t; i < 16 * WP; i += 256) {
        int kx = i / WP, w = i - kx * WP;
        int m = (kx * w) % WP;                        // exact angle reduction
        float a = -TWO_PI * (float)m * (1.0f / WP);
        float s, c; __sincosf(a, &s, &c);
        twr[i] = c; twi[i] = s;
    }
    int wid = t >> 5, lane = t & 31;
    size_t row = (size_t)blockIdx.x * 8 + wid;
    const __bf16* g = h + row * WP;                   // 528B, 16B aligned
    unsigned lbase = (unsigned)(size_t)&rowbuf[wid][0];
    async_b128(lbase + lane * 16, g + lane * 8);      // 512B
    if (lane == 0) async_b128(lbase + 512, g + 256);  // tail 16B
    wait_async0();
    __syncthreads();
    int kx = lane & 15, half = lane >> 4;
    float re = 0.f, im = 0.f;
    const float* cr = &twr[kx * WP];
    const float* ci = &twi[kx * WP];
    for (int w = half * 132; w < half * 132 + 132; ++w) {
        float v = (float)rowbuf[wid][w];
        re += v * cr[w]; im += v * ci[w];
    }
    re += __shfl_xor(re, 16, 32);
    im += __shfl_xor(im, 16, 32);
    if (lane < 16) {
        X1[(row * 16 + kx) * 2 + 0] = re;
        X1[(row * 16 + kx) * 2 + 1] = im;
    }
}

// ---------------- forward partial DFT along H (32 ky modes) ------------------
__global__ __launch_bounds__(256)
void k_dfth(const float* __restrict__ X1, float* __restrict__ Xf) {
    __shared__ float er[16 * HP];
    __shared__ float ei[16 * HP];
    int half = blockIdx.x, bc = blockIdx.y, t = threadIdx.x;
    for (int i = t; i < 16 * HP; i += 256) {
        int kyl = i / HP, hh = i - kyl * HP;
        int kyact = half ? (kyl + 248) : kyl;
        int m = (kyact * hh) % HP;
        float a = -TWO_PI * (float)m * (1.0f / HP);
        float s, c; __sincosf(a, &s, &c);
        er[i] = c; ei[i] = s;
    }
    __syncthreads();
    int kyl = t >> 4, kx = t & 15;
    const float* cr = &er[kyl * HP];
    const float* ci = &ei[kyl * HP];
    const float* xp = X1 + ((size_t)bc * HP * 16 + kx) * 2;
    float re = 0.f, im = 0.f;
    for (int hh = 0; hh < HP; ++hh) {
        float xr = xp[hh * 32 + 0], xi = xp[hh * 32 + 1];
        float c = cr[hh], s = ci[hh];
        re += xr * c - xi * s;
        im += xr * s + xi * c;
    }
    int ky = half * 16 + kyl;
    size_t o = (((size_t)bc * NMODE_Y + ky) * NMODE_X + kx) * 2;
    Xf[o] = re; Xf[o + 1] = im;
}

// ---------------- per-mode complex channel mix, bf16 WMMA --------------------
// D(16x128) = A(16x128) * B(128x128): A=[Re x|Im x], B=[[Wr,Wi],[-Wi,Wr]].
__global__ __launch_bounds__(256)
void k_mix(const float* __restrict__ Xf, const float* __restrict__ w1,
           const float* __restrict__ w2, float* __restrict__ Mx) {
    __shared__ __align__(32) __bf16 Bt[8 * 2048];    // 8 subtiles (16n x 128k)
    __shared__ __align__(32) __bf16 Al[2048];        // 1 subtile  (16m x 128k)
    int mode = blockIdx.x;
    int ky = mode >> 4, kx = mode & 15;
    int t = threadIdx.x;
    const float* wp; int kyp;
    if (ky < 16) { wp = w1; kyp = ky; } else { wp = w2; kyp = ky - 16; }
    for (int e = t; e < 128 * 128; e += 256) {
        int n = e >> 7, k = e & 127;
        int ci = k & 63, hk = k >> 6;
        int co = n & 63, hn = n >> 6;
        size_t wi = (size_t)ci * 32768 + (size_t)co * 512 + (size_t)kyp * 32 + kx * 2;
        float val;
        if (hn == 0) val = (hk == 0) ? wp[wi] : -wp[wi + 1];
        else         val = (hk == 0) ? wp[wi + 1] : wp[wi];
        Bt[(n >> 4) * 2048 + swz_idx(n & 15, k)] = (__bf16)val;
    }
    for (int e = t; e < 16 * 128; e += 256) {
        int bb = e >> 7, k = e & 127;
        int ci = k & 63, hk = k >> 6;
        size_t xi = (((size_t)(bb * CHN + ci) * NMODE_Y + ky) * NMODE_X + kx) * 2 + hk;
        Al[swz_idx(bb, k)] = (__bf16)Xf[xi];
    }
    __syncthreads();
    int wid = t >> 5, lane = t & 31;
    v8f acc = {0.f, 0.f, 0.f, 0.f, 0.f, 0.f, 0.f, 0.f};
    for (int ks = 0; ks < 4; ++ks) {
        v16bf a = load_frag_swz(Al, ks, lane);
        v16bf b = load_frag_swz(Bt + wid * 2048, ks, lane);
        acc = __builtin_amdgcn_wmma_f32_16x16x32_bf16(false, a, false, b,
                                                      (short)0, acc, false, false);
    }
    int hi = lane >> 4, nl = lane & 15;
    for (int r = 0; r < 8; ++r) {
        int bb = r + hi * 8;
        int n = wid * 16 + nl;
        int co = n & 63, reim = n >> 6;
        size_t o = (((size_t)(bb * CHN + co) * NMODE_Y + ky) * NMODE_X + kx) * 2 + reim;
        Mx[o] = acc[r];
    }
}

// ---------------- inverse partial DFT along H --------------------------------
__global__ __launch_bounds__(256)
void k_idfth(const float* __restrict__ Mx, float* __restrict__ Y1) {
    __shared__ __align__(16) float mlds[NMODE_Y * NMODE_X * 2];  // 4KB
    int bc = blockIdx.x, t = threadIdx.x;
    const float* g = Mx + (size_t)bc * (NMODE_Y * NMODE_X * 2);
    unsigned lbase = (unsigned)(size_t)mlds;
    async_b128(lbase + t * 16, g + t * 4);            // 256 * 16B = 4KB
    wait_async0();
    __syncthreads();
    for (int o = t; o < HP * NMODE_X; o += 256) {
        int hh = o >> 4, kx = o & 15;
        float re = 0.f, im = 0.f;
        for (int ky = 0; ky < NMODE_Y; ++ky) {
            int kyact = (ky < 16) ? ky : (ky + 232);
            int m = (kyact * hh) % HP;
            float a = TWO_PI * (float)m * (1.0f / HP);
            float s, c; __sincosf(a, &s, &c);
            float xr = mlds[(ky * 16 + kx) * 2 + 0];
            float xi = mlds[(ky * 16 + kx) * 2 + 1];
            re += xr * c - xi * s;
            im += xr * s + xi * c;
        }
        size_t oo = (((size_t)bc * HP + hh) * 16 + kx) * 2;
        Y1[oo] = re; Y1[oo + 1] = im;
    }
}

// ---------------- inverse partial DFT along W (real bf16 out) ----------------
__global__ __launch_bounds__(256)
void k_idftw(const float* __restrict__ Y1, __bf16* __restrict__ sp) {
    __shared__ float twr[16 * WP];
    __shared__ float twi[16 * WP];
    __shared__ float yb[8][32];
    int t = threadIdx.x;
    for (int i = t; i < 16 * WP; i += 256) {
        int kx = i / WP, w = i - kx * WP;
        int m = (kx * w) % WP;
        float a = TWO_PI * (float)m * (1.0f / WP);
        float s, c; __sincosf(a, &s, &c);
        twr[i] = c; twi[i] = s;
    }
    int wid = t >> 5, lane = t & 31;
    size_t row = (size_t)blockIdx.x * 8 + wid;
    if (lane < 16) {
        float fr = Y1[(row * 16 + lane) * 2 + 0];
        float fi = Y1[(row * 16 + lane) * 2 + 1];
        float cc = ((lane == 0) ? 1.0f : 2.0f) * (1.0f / (float)NP);  // irfft2 norm
        yb[wid][lane * 2 + 0] = fr * cc;
        yb[wid][lane * 2 + 1] = fi * cc;
    }
    __syncthreads();
    __bf16* orow = sp + row * WP;
    for (int w = lane; w < WP; w += 32) {
        float acc = 0.f;
        for (int kx = 0; kx < 16; ++kx) {
            float c = twr[kx * WP + w], s = twi[kx * WP + w];
            acc += yb[wid][kx * 2] * c - yb[wid][kx * 2 + 1] * s;
        }
        orow[w] = (__bf16)acc;
    }
}

// ---------------- 1x1 conv / decoder GEMM: TDM + WMMA ------------------------
// out[64co][128px] tile; activation tile staged by the Tensor Data Mover
// (2D D#: 128 x 64 bf16, stride NP), then swizzled to fragment-major LDS.
__global__ __launch_bounds__(256)
void k_pconv(const __bf16* __restrict__ hin, const float* __restrict__ w,
             const float* __restrict__ bias, __bf16* __restrict__ out,
             int np, int crop, int dogelu) {
    __shared__ __align__(32) __bf16 Wl[4 * 1024];     // 4 subtiles (16co x 64ci)
    __shared__ __align__(32) __bf16 Ht[8 * 1024];     // 8 subtiles (16px x 64ci)
    __shared__ float bl[64];
#ifdef HAVE_TDM
    __shared__ __align__(32) __bf16 HtRM[64 * 128];   // row-major TDM landing
#endif
    int t = threadIdx.x;
    int b = blockIdx.y;
    int px0 = blockIdx.x * 128;
    for (int e = t; e < 64 * 64; e += 256) {
        int co = e >> 6, ci = e & 63;
        Wl[(co >> 4) * 1024 + swz_idx(co & 15, ci)] = (__bf16)w[e];
    }
    if (t < 64) bl[t] = bias[t];

    int soff = crop ? ((px0 >> 8) * WP + (px0 & 255)) : px0;
    size_t cstride = crop ? (size_t)NP : (size_t)np;
    const __bf16* gsrc = hin + (size_t)b * CHN * cstride + soff;

#ifdef HAVE_TDM
    if (t < 32) {                                     // wave 0 drives the TDM
        unsigned lds_addr = (unsigned)(size_t)HtRM;
        unsigned long long ga = (unsigned long long)(size_t)gsrc;
        u32x4 g0;
        g0.x = 1u;                                    // count=1 valid user D#
        g0.y = lds_addr;                              // LDS dest (bytes)
        g0.z = (unsigned)(ga & 0xffffffffull);        // global_addr[31:0]
        g0.w = (unsigned)((ga >> 32) & 0x01ffffffull) | (2u << 30);  // type=2
        i32x8 g1;
        g1[0] = (1 << 16);                            // data_size=1 (2 bytes)
        g1[1] = (int)((NP & 0xffff) << 16);           // tensor_dim0[15:0]
        g1[2] = (int)((NP >> 16) | (64 << 16));       // td0[31:16] | td1[15:0]
        g1[3] = (int)(128 << 16);                     // td1 hi=0 | tile_dim0=128
        g1[4] = 64;                                   // tile_dim1=64, tile_dim2=0
        g1[5] = (int)NP;                              // tensor_dim0_stride[31:0]
        g1[6] = 0;                                    // strides hi
        g1[7] = 0;
        i32x4 z4 = {0, 0, 0, 0};                      // D# groups 2/3: 2D tensor
        i32x8 z8 = {0, 0, 0, 0, 0, 0, 0, 0};          // extended group (unused)
        __builtin_amdgcn_tensor_load_to_lds(g0, g1, z4, z4, z8, 0);
        wait_tensor0();
    }
    __syncthreads();
    // swizzle row-major [ci][px] -> fragment-major [px-subtile][...]
    for (int e = t; e < 64 * 128; e += 256) {
        int ci = e >> 7, pxl = e & 127;
        __bf16 v = HtRM[ci * 128 + pxl];
        if (!crop && (px0 + pxl >= np)) v = (__bf16)0.0f;  // belt & suspenders
        Ht[(pxl >> 4) * 1024 + swz_idx(pxl & 15, ci)] = v;
    }
#else
    {
        int ci = t >> 2, blk = t & 3;
        const __bf16* g = gsrc + (size_t)ci * cstride - soff;
        for (int i2 = 0; i2 < 32; ++i2) {
            int pxl = blk * 32 + i2;
            int p = px0 + pxl;
            __bf16 v = (__bf16)0.0f;
            if (p < np) {
                if (crop) { int y = p >> 8, xx = p & 255; v = g[y * WP + xx]; }
                else      { v = g[p]; }
            }
            Ht[(pxl >> 4) * 1024 + swz_idx(pxl & 15, ci)] = v;
        }
    }
#endif
    __syncthreads();

    int wid = t >> 5, lane = t & 31;
    int cot = wid >> 1, ph = wid & 1;
    int hi = lane >> 4, nl = lane & 15;
    for (int n = 0; n < 4; ++n) {
        v8f acc = {0.f, 0.f, 0.f, 0.f, 0.f, 0.f, 0.f, 0.f};
        for (int ks = 0; ks < 2; ++ks) {
            v16bf a = load_frag_swz(Wl + cot * 1024, ks, lane);
            v16bf bb = load_frag_swz(Ht + (ph * 4 + n) * 1024, ks, lane);
            acc = __builtin_amdgcn_wmma_f32_16x16x32_bf16(false, a, false, bb,
                                                          (short)0, acc, false, false);
        }
        int px = px0 + ph * 64 + n * 16 + nl;
        if (px < np) {
            for (int r = 0; r < 8; ++r) {
                int co = cot * 16 + r + hi * 8;
                float v = acc[r] + bl[co];
                if (dogelu) v = gelu_f(v);
                out[((size_t)b * CHN + co) * np + px] = (__bf16)v;
            }
        }
    }
}

// ---------------- group-norm: deterministic two-stage reduction --------------
__global__ __launch_bounds__(256)
void k_gnpart(const __bf16* __restrict__ sp, float* __restrict__ part) {
    int cl = blockIdx.x, g = blockIdx.y, b = blockIdx.z, t = threadIdx.x;
    int c = g * 16 + cl;
    const __bf16* p = sp + ((size_t)b * CHN + c) * NP;
    float s = 0.f, q = 0.f;
    for (int i = t * 8; i < NP; i += 2048) {
        B16x8 v8; v8.u = *(const uint4*)(p + i);
        #pragma unroll
        for (int j = 0; j < 8; ++j) {
            float v = (float)v8.h[j];
            s += v; q += v * v;
        }
    }
    __shared__ float sh[256];
    __shared__ float sh2[256];
    sh[t] = s; sh2[t] = q;
    __syncthreads();
    for (int st = 128; st > 0; st >>= 1) {
        if (t < st) { sh[t] += sh[t + st]; sh2[t] += sh2[t + st]; }
        __syncthreads();
    }
    if (t == 0) {
        int o = ((b * 4 + g) * 16 + cl) * 2;
        part[o] = sh[0]; part[o + 1] = sh2[0];
    }
}

__global__ void k_gnfin(const float* __restrict__ part, float* __restrict__ stats) {
    int i = threadIdx.x;                       // 64 threads = (b,g)
    if (i < 64) {
        float s = 0.f, q = 0.f;
        for (int cl = 0; cl < 16; ++cl) {
            s += part[(i * 16 + cl) * 2];
            q += part[(i * 16 + cl) * 2 + 1];
        }
        float n = 16.0f * (float)NP;
        float mu = s / n;
        float var = q / n - mu * mu;
        stats[i * 2] = mu;
        stats[i * 2 + 1] = rsqrtf(var + 1e-5f);
    }
}

// ---------------- fused normalize + residual + gelu, 8-wide ------------------
__global__ __launch_bounds__(256)
void k_apply(const __bf16* __restrict__ sp, const __bf16* __restrict__ cv,
             const float* __restrict__ stats, const float* __restrict__ gg,
             const float* __restrict__ gb, __bf16* __restrict__ hout, int dogelu) {
    size_t idx = (size_t)blockIdx.x * 256 + threadIdx.x;
    size_t i8 = idx * 8;
    size_t t = i8 / NP;
    int c = (int)(t % CHN);
    int b = (int)(t / CHN);
    int g = c >> 4;
    float mu = stats[(b * 4 + g) * 2];
    float is = stats[(b * 4 + g) * 2 + 1];
    float ga = gg[c], be = gb[c];
    B16x8 vs, vc, vo;
    vs.u = *(const uint4*)(sp + i8);
    vc.u = *(const uint4*)(cv + i8);
    #pragma unroll
    for (int j = 0; j < 8; ++j) {
        float v = ((float)vs.h[j] - mu) * is * ga + be + (float)vc.h[j];
        if (dogelu) v = gelu_f(v);
        vo.h[j] = (__bf16)v;
    }
    *(uint4*)(hout + i8) = vo.u;
}

// ---------------- final 64 -> 1 projection (f32 out), 8-wide -----------------
__global__ __launch_bounds__(256)
void k_dec2(const __bf16* __restrict__ d1, const float* __restrict__ w2,
            const float* __restrict__ b2, float* __restrict__ out) {
    size_t idx = (size_t)blockIdx.x * 256 + threadIdx.x;  // B*65536/8 items
    int p8 = (int)(idx & 8191) * 8;
    int b  = (int)(idx >> 13);
    const __bf16* base = d1 + (size_t)b * CHN * (H0 * W0) + p8;
    float acc[8];
    float bv = b2[0];
    #pragma unroll
    for (int j = 0; j < 8; ++j) acc[j] = bv;
    for (int ci = 0; ci < CHN; ++ci) {
        if (ci + 8 < CHN) __builtin_prefetch(base + (size_t)(ci + 8) * (H0 * W0), 0, 1);
        B16x8 v8; v8.u = *(const uint4*)(base + (size_t)ci * (H0 * W0));
        float wv = w2[ci];
        #pragma unroll
        for (int j = 0; j < 8; ++j) acc[j] += wv * (float)v8.h[j];
    }
    float* o = out + (size_t)b * (H0 * W0) + p8;
    #pragma unroll
    for (int j = 0; j < 8; ++j) o[j] = acc[j];
}

// ---------------- host driver -----------------------------------------------
extern "C" void kernel_launch(void* const* d_in, const int* in_sizes, int n_in,
                              void* d_out, int out_size, void* d_ws, size_t ws_size,
                              hipStream_t stream) {
    (void)in_sizes; (void)n_in; (void)out_size; (void)ws_size;
    const float* x      = (const float*)d_in[0];
    const float* lift_w = (const float*)d_in[1];
    const float* lift_b = (const float*)d_in[2];
    const float* sp_w1  = (const float*)d_in[3];
    const float* sp_w2  = (const float*)d_in[4];
    const float* conv_w = (const float*)d_in[5];
    const float* conv_b = (const float*)d_in[6];
    const float* gn_g   = (const float*)d_in[7];
    const float* gn_b   = (const float*)d_in[8];
    const float* dec_w1 = (const float*)d_in[9];
    const float* dec_b1 = (const float*)d_in[10];
    const float* dec_w2 = (const float*)d_in[11];
    const float* dec_b2 = (const float*)d_in[12];
    float* outp = (float*)d_out;

    const size_t SZ_ACT = (size_t)BATCH * CHN * NP * 2;            // bf16 act
    const size_t SZ_X1  = (size_t)BATCH * CHN * HP * 16 * 2 * 4;   // cplx f32
    const size_t SZ_XF  = (size_t)BATCH * CHN * NMODE_Y * NMODE_X * 2 * 4;

    char* ws = (char*)d_ws;
    __bf16* h    = (__bf16*)(ws);
    __bf16* sp   = (__bf16*)(ws + SZ_ACT);
    __bf16* cv   = (__bf16*)(ws + 2 * SZ_ACT);     // conv out; reused as dec1
    float*  X1   = (float*)(ws + 3 * SZ_ACT);
    float*  Y1   = (float*)(ws + 3 * SZ_ACT + SZ_X1);
    float*  Xf   = (float*)(ws + 3 * SZ_ACT + 2 * SZ_X1);
    float*  Mx   = (float*)(ws + 3 * SZ_ACT + 2 * SZ_X1 + SZ_XF);
    float*  part = (float*)(ws + 3 * SZ_ACT + 2 * SZ_X1 + 2 * SZ_XF);
    float*  stats= (float*)(ws + 3 * SZ_ACT + 2 * SZ_X1 + 2 * SZ_XF + 8192);

    const int nVec8 = (int)(((size_t)BATCH * CHN * NP) / 8 / 256); // 34848
    const int nRows = (BATCH * CHN * HP) / 8;                      // 33792
    const int nConvTiles = (NP + 127) / 128;                       // 545
    const int nDecTiles  = (H0 * W0) / 128;                        // 512

    k_lift<<<nVec8, 256, 0, stream>>>(x, lift_w, lift_b, h);

    for (int k = 0; k < NLAYER; ++k) {
        const float* w1 = sp_w1 + (size_t)k * CHN * CHN * 16 * 16 * 2;
        const float* w2 = sp_w2 + (size_t)k * CHN * CHN * 16 * 16 * 2;
        const float* cw = conv_w + (size_t)k * CHN * CHN;
        const float* cb = conv_b + (size_t)k * CHN;
        const float* gg = gn_g + (size_t)k * CHN;
        const float* gb = gn_b + (size_t)k * CHN;

        k_dftw <<<nRows, 256, 0, stream>>>(h, X1);
        k_dfth <<<dim3(2, BATCH * CHN), 256, 0, stream>>>(X1, Xf);
        k_mix  <<<NMODE_Y * NMODE_X, 256, 0, stream>>>(Xf, w1, w2, Mx);
        k_idfth<<<BATCH * CHN, 256, 0, stream>>>(Mx, Y1);
        k_idftw<<<nRows, 256, 0, stream>>>(Y1, sp);
        k_pconv<<<dim3(nConvTiles, BATCH), 256, 0, stream>>>(h, cw, cb, cv,
                                                             NP, 0, 0);
        k_gnpart<<<dim3(16, 4, BATCH), 256, 0, stream>>>(sp, part);
        k_gnfin<<<1, 64, 0, stream>>>(part, stats);
        k_apply<<<nVec8, 256, 0, stream>>>(sp, cv, stats, gg, gb, h,
                                           (k < NLAYER - 1) ? 1 : 0);
    }

    // decoder: crop + 64->64 conv + gelu (WMMA), then 64->1 projection (f32)
    k_pconv<<<dim3(nDecTiles, BATCH), 256, 0, stream>>>(h, dec_w1, dec_b1,
                                                        cv, H0 * W0, 1, 1);
    k_dec2<<<((BATCH * H0 * W0) / 8) / 256, 256, 0, stream>>>(cv, dec_w2, dec_b2, outp);
}
